// JPANet_52991306498490
// MI455X (gfx1250) — matrix operand order repaired and checked by reading, not compiled, not run on hardware
//
#include <hip/hip_runtime.h>
#include <hip/hip_bf16.h>
#include <cmath>

// ---------------------------------------------------------------------------
// JPANet fused forward for MI455X (gfx1250, wave32, WMMA 16x16x32 f16)
// ---------------------------------------------------------------------------

typedef __attribute__((ext_vector_type(16))) _Float16 v16h;
typedef __attribute__((ext_vector_type(8)))  float    v8f;
typedef __attribute__((ext_vector_type(4)))  unsigned int v4u;

#define BATCH 16384
#define NCHUNK 8
#define BCHUNK (BATCH / NCHUNK)

__constant__ float c_mu_mean[3] = {65.681f, 5.0793e-05f, -0.0017228f};
__constant__ float c_mu_std [3] = {38.2836f, 1.8144f, 1.12f};
__constant__ float c_nu_mean[3] = {66.7863f, 0.141f, 0.7417f};
__constant__ float c_nu_std [3] = {66.7836f, 1.7177f, 1.2575f};
__constant__ float c_jet_mean[6] = {63.964f, -0.0047165f, -0.0023042f, 0.36918f, 0.50398f, 0.42594f};
__constant__ float c_jet_std [6] = {50.1006f, 1.8142f, 1.6456f, 0.4087f, 0.3379f, 0.3796f};
__constant__ float c_lm_mean[36] = {
    4.512f, 5.4f,5.4f,5.4f,5.4f,5.4f,5.4f,5.4f,
    2.277f, 4.857f,4.857f,4.857f,4.857f,4.857f,4.857f,
    2.277f, 4.857f,4.857f,4.857f,4.857f,4.857f,
    2.277f, 4.857f,4.857f,4.857f,4.857f,
    2.277f, 4.857f,4.857f,4.857f,
    2.277f, 4.857f,4.857f,
    2.277f, 4.857f,
    2.277f};
__constant__ float c_lm_std[36] = {
    0.2378f, 0.483f,0.483f,0.483f,0.483f,0.483f,0.483f,0.483f,
    0.431f, 0.7387f,0.7387f,0.7387f,0.7387f,0.7387f,0.7387f,
    0.431f, 0.7387f,0.7387f,0.7387f,0.7387f,0.7387f,
    0.431f, 0.7387f,0.7387f,0.7387f,0.7387f,
    0.431f, 0.7387f,0.7387f,0.7387f,
    0.431f, 0.7387f,0.7387f,
    0.431f, 0.7387f,
    0.431f};

__device__ __forceinline__ float siluf(float x) { return x / (1.f + __expf(-x)); }

// ---------------------------------------------------------------------------
// Weight fp32 -> fp16 conversion (once per launch; weights are L2-resident)
// ---------------------------------------------------------------------------
__global__ __launch_bounds__(256)
void k_cvt(const float* __restrict__ s, _Float16* __restrict__ d, int n)
{
    const int i = blockIdx.x * blockDim.x + threadIdx.x;
    if (i < n) d[i] = (_Float16)s[i];
}

// ---------------------------------------------------------------------------
// k_embed: normalization + tiny first layers (mu/nu 3->64->128, mass 36->128,
// jet 6->128).  One block (128 threads) per batch element.
// ---------------------------------------------------------------------------
__global__ __launch_bounds__(128)
void k_embed(const float* __restrict__ mu, const float* __restrict__ nu,
             const float* __restrict__ jet, const float* __restrict__ masses,
             const float* __restrict__ mu1w, const float* __restrict__ mu1b,
             const float* __restrict__ mu2w, const float* __restrict__ mu2b,
             const float* __restrict__ nu1w, const float* __restrict__ nu1b,
             const float* __restrict__ nu2w, const float* __restrict__ nu2b,
             const float* __restrict__ m1w,  const float* __restrict__ m1b,
             const float* __restrict__ j1w,  const float* __restrict__ j1b,
             float* __restrict__ evin, float* __restrict__ mass_h,
             float* __restrict__ jet_h)
{
    const int b = blockIdx.x;
    const int t = threadIdx.x;
    __shared__ float hmu[64], hnu[64], mz[36], jz[42];

    if (t < 64) {
        float a = mu1b[t];
        #pragma unroll
        for (int k = 0; k < 3; ++k)
            a += ((mu[(size_t)b*3 + k] - c_mu_mean[k]) / c_mu_std[k]) * mu1w[t*3 + k];
        hmu[t] = siluf(a);
    } else {
        const int u = t - 64;
        float a = nu1b[u];
        #pragma unroll
        for (int k = 0; k < 3; ++k)
            a += ((nu[(size_t)b*3 + k] - c_nu_mean[k]) / c_nu_std[k]) * nu1w[u*3 + k];
        hnu[u] = siluf(a);
    }
    if (t < 36)
        mz[t] = (log1pf(masses[(size_t)b*36 + t]) - c_lm_mean[t]) / c_lm_std[t];
    if (t >= 64 && t < 106) {
        const int u = t - 64, k = u % 6;
        jz[u] = (jet[(size_t)b*42 + u] - c_jet_mean[k]) / c_jet_std[k];
    }
    __syncthreads();

    {
        float a = mu2b[t], c = nu2b[t];
        #pragma unroll 8
        for (int k = 0; k < 64; ++k) {
            a += hmu[k] * mu2w[t*64 + k];
            c += hnu[k] * nu2w[t*64 + k];
        }
        evin[(size_t)b*256 + t]       = siluf(a);
        evin[(size_t)b*256 + 128 + t] = siluf(c);
    }
    {
        float a = m1b[t];
        #pragma unroll
        for (int k = 0; k < 36; ++k) a += mz[k] * m1w[t*36 + k];
        mass_h[(size_t)b*128 + t] = siluf(a);
    }
    for (int tt = 0; tt < 7; ++tt) {
        float a = j1b[t];
        #pragma unroll
        for (int k = 0; k < 6; ++k) a += jz[tt*6 + k] * j1w[t*6 + k];
        jet_h[((size_t)b*7 + tt)*128 + t] = siluf(a);
    }
}

// ---------------------------------------------------------------------------
// Generic WMMA GEMM:  out[m,:] = act( X[m,:K] @ W[N,K]^T + bias )
// 128 threads = 4 waves; each wave owns TWO 16-row M-tiles (block = 128 rows).
// A fragments loaded once per block directly from global (float4 + cvt) into
// persistent registers; each B fragment (two global_load_b128 of f16 weights)
// feeds TWO independent WMMA accumulation chains (latency hiding + 2x reuse).
// Output row scatter: off(m,n) = (m/TOKT)*bstride + (m%TOKT + tokOff)*tokStride + n
// ACT: 0 = none, 1 = SiLU, 2 = SiLU + nan_to_num
// ---------------------------------------------------------------------------
template<int K, int ACT, bool OUTH, int TOKT>
__global__ __launch_bounds__(128)
void k_gemm(const float* __restrict__ X, const _Float16* __restrict__ Wh,
            const float* __restrict__ bias, int N,
            float* __restrict__ outF, _Float16* __restrict__ outH,
            int tokOff, int tokStride, int bstride)
{
    const int tid  = threadIdx.x;
    const int wave = tid >> 5, lane = tid & 31;
    const int g = lane >> 4, l16 = lane & 15;
    const size_t rowBase = (size_t)blockIdx.x * 128 + (size_t)wave * 32;

    // A fragments: CDNA5 16x32 f16 layout = two runs of 8 consecutive halves
    union AF { v16h v; _Float16 h[16]; v4u q[2]; };
    AF a[2][K / 32];
    #pragma unroll
    for (int t = 0; t < 2; ++t) {
        const float* xr = X + (rowBase + t*16 + l16) * K;
        #pragma unroll
        for (int ks = 0; ks < K/32; ++ks) {
            const int k0 = ks * 32;
            const float4 x0 = *(const float4*)(xr + k0 + 8*g);
            const float4 x1 = *(const float4*)(xr + k0 + 8*g + 4);
            const float4 x2 = *(const float4*)(xr + k0 + 16 + 8*g);
            const float4 x3 = *(const float4*)(xr + k0 + 16 + 8*g + 4);
            AF& f = a[t][ks];
            f.h[0]=(_Float16)x0.x; f.h[1]=(_Float16)x0.y; f.h[2]=(_Float16)x0.z; f.h[3]=(_Float16)x0.w;
            f.h[4]=(_Float16)x1.x; f.h[5]=(_Float16)x1.y; f.h[6]=(_Float16)x1.z; f.h[7]=(_Float16)x1.w;
            f.h[8]=(_Float16)x2.x; f.h[9]=(_Float16)x2.y; f.h[10]=(_Float16)x2.z; f.h[11]=(_Float16)x2.w;
            f.h[12]=(_Float16)x3.x; f.h[13]=(_Float16)x3.y; f.h[14]=(_Float16)x3.z; f.h[15]=(_Float16)x3.w;
        }
    }

    for (int n0 = 0; n0 < N; n0 += 16) {
        union { v8f v; float f[8]; } acc0, acc1;
        #pragma unroll
        for (int r = 0; r < 8; ++r) { acc0.f[r] = 0.f; acc1.f[r] = 0.f; }

        const _Float16* wbase = Wh + (size_t)(n0 + l16) * K + 8*g;
        __builtin_prefetch(wbase + (size_t)16 * K, 0, 1);   // next n-tile row

        #pragma unroll
        for (int ks = 0; ks < K/32; ++ks) {
            AF bf;
            const _Float16* wp = wbase + ks*32;
            bf.q[0] = *(const v4u*)(wp);
            bf.q[1] = *(const v4u*)(wp + 16);
            acc0.v = __builtin_amdgcn_wmma_f32_16x16x32_f16(
                         false, a[0][ks].v, false, bf.v, (short)0, acc0.v, false, false);
            acc1.v = __builtin_amdgcn_wmma_f32_16x16x32_f16(
                         false, a[1][ks].v, false, bf.v, (short)0, acc1.v, false, false);
        }

        const int n = n0 + l16;
        const float bv = bias[n];
        #pragma unroll
        for (int t = 0; t < 2; ++t) {
            #pragma unroll
            for (int r = 0; r < 8; ++r) {
                const size_t gm = rowBase + t*16 + r + 8*g;
                float x = (t == 0 ? acc0.f[r] : acc1.f[r]) + bv;
                if (ACT >= 1) x = x / (1.f + __expf(-x));
                if (ACT == 2 && !(x == x)) x = 0.f;
                const size_t off = (gm / (size_t)TOKT) * (size_t)bstride +
                                   (size_t)((int)(gm % (size_t)TOKT) + tokOff) * (size_t)tokStride +
                                   (size_t)n;
                if (OUTH) outH[off] = (_Float16)x;
                else      outF[off] = x;
            }
        }
    }
}

// ---------------------------------------------------------------------------
// Attention context: one thread per (batch, head, query-row).
// Builds symmetric mass-bias (vec_to_sym) + pad mask, softmax, context.
// T=7 (jet_attn, bias 28 tri) or T=8 (all_attn, bias 36 tri, token0 unpadded).
// ---------------------------------------------------------------------------
template<int T>
__global__ __launch_bounds__(256)
void k_attn(const _Float16* __restrict__ qkv, const float* __restrict__ biasRaw,
            const float* __restrict__ jetRaw, float* __restrict__ ctx,
            int c0, int Bc)
{
    const int idx = blockIdx.x * blockDim.x + threadIdx.x;
    if (idx >= Bc * 2 * T) return;
    const int bl  = idx / (2 * T);
    const int rem = idx - bl * 2 * T;
    const int h = rem / T;
    const int i = rem % T;
    const int b = c0 + bl;

    bool pad[T];
    #pragma unroll
    for (int j = 0; j < T; ++j) {
        if (T == 8 && j == 0) { pad[j] = false; continue; }
        const int jj = (T == 8) ? j - 1 : j;
        const float v = (jetRaw[(size_t)b*42 + jj*6] - 63.964f) / 50.1006f;
        pad[j] = (v == 0.f);
    }

    const float* brow = biasRaw + (size_t)b * (T * (T + 1) / 2);

    union H8 { v4u q; _Float16 h[8]; };
    const _Float16* qp = qkv + ((size_t)(bl*T + i) * 768 + h*128);
    H8 qb[16];
    #pragma unroll
    for (int c = 0; c < 16; ++c) qb[c].q = *(const v4u*)(qp + c*8);

    float s[T];
    #pragma unroll
    for (int j = 0; j < T; ++j) {
        const int lo = (i < j) ? i : j, hi = (i < j) ? j : i;
        const int off = lo * T - (lo * (lo - 1)) / 2 + (hi - lo);
        const float bias = (pad[i] || pad[j]) ? -INFINITY : brow[off];
        const _Float16* kp = qkv + ((size_t)(bl*T + j) * 768 + 256 + h*128);
        float d = 0.f;
        #pragma unroll
        for (int c = 0; c < 16; ++c) {
            H8 kb; kb.q = *(const v4u*)(kp + c*8);
            #pragma unroll
            for (int u = 0; u < 8; ++u) d += (float)qb[c].h[u] * (float)kb.h[u];
        }
        s[j] = d * 0.08838834764831845f + bias;
    }
    float m = s[0];
    #pragma unroll
    for (int j = 1; j < T; ++j) m = fmaxf(m, s[j]);
    float sum = 0.f;
    #pragma unroll
    for (int j = 0; j < T; ++j) { s[j] = __expf(s[j] - m); sum += s[j]; }
    const float inv = 1.f / sum;

    float* cptr = ctx + ((size_t)(bl*T + i) * 256 + h*128);
    const _Float16* vbase = qkv + ((size_t)(bl*T) * 768 + 512 + h*128);
    #pragma unroll 2
    for (int c = 0; c < 16; ++c) {
        float acc[8];
        #pragma unroll
        for (int u = 0; u < 8; ++u) acc[u] = 0.f;
        #pragma unroll
        for (int j = 0; j < T; ++j) {
            H8 vb; vb.q = *(const v4u*)(vbase + (size_t)j*768 + c*8);
            #pragma unroll
            for (int u = 0; u < 8; ++u) acc[u] += s[j] * (float)vb.h[u];
        }
        #pragma unroll
        for (int u = 0; u < 8; ++u) cptr[c*8 + u] = acc[u] * inv;
    }
}

// ---------------------------------------------------------------------------
// LayerNorm over 256 features, in place. One block (256 threads) per row.
// ---------------------------------------------------------------------------
__global__ __launch_bounds__(256)
void k_lnorm(float* __restrict__ x, const float* __restrict__ g,
             const float* __restrict__ bta)
{
    const int bb = blockIdx.x, d = threadIdx.x;
    __shared__ float red[256];
    const float v = x[(size_t)bb*256 + d];
    red[d] = v; __syncthreads();
    for (int s = 128; s > 0; s >>= 1) { if (d < s) red[d] += red[d + s]; __syncthreads(); }
    const float mean = red[0] / 256.f; __syncthreads();
    const float dv = v - mean;
    red[d] = dv * dv; __syncthreads();
    for (int s = 128; s > 0; s >>= 1) { if (d < s) red[d] += red[d + s]; __syncthreads(); }
    const float var = red[0] / 256.f;
    x[(size_t)bb*256 + d] = dv * rsqrtf(var + 1e-5f) * g[d] + bta[d];
}

// ---------------------------------------------------------------------------
// mass_linear1 (256->28) and mass_linear2 (256->36). One block (64) per row.
// ---------------------------------------------------------------------------
__global__ __launch_bounds__(64)
void k_masslin(const float* __restrict__ mass,
               const float* __restrict__ w1, const float* __restrict__ b1,
               const float* __restrict__ w2, const float* __restrict__ b2,
               float* __restrict__ jm, float* __restrict__ tm)
{
    const int b = blockIdx.x, o = threadIdx.x;
    const float* x = mass + (size_t)b * 256;
    if (o < 28) {
        const float* w = w1 + (size_t)o * 256;
        float a = b1[o];
        #pragma unroll 8
        for (int k = 0; k < 256; ++k) a += x[k] * w[k];
        jm[(size_t)b*28 + o] = a;
    } else {
        const int u = o - 28;
        if (u < 36) {
            const float* w = w2 + (size_t)u * 256;
            float a = b2[u];
            #pragma unroll 8
            for (int k = 0; k < 256; ++k) a += x[k] * w[k];
            tm[(size_t)b*36 + u] = a;
        }
    }
}

// ---------------------------------------------------------------------------
// Head: pooled softmax attention over 8 tokens, post_pool 256->128 SiLU,
// out 128->3, log_softmax.  One block (256 threads) per batch element.
// ---------------------------------------------------------------------------
__global__ __launch_bounds__(256)
void k_head(const float* __restrict__ total, const float* __restrict__ jetRaw,
            const float* __restrict__ poolw, const float* __restrict__ poolb,
            const float* __restrict__ ppw, const float* __restrict__ ppb,
            const float* __restrict__ ow, const float* __restrict__ ob,
            float* __restrict__ out)
{
    const int b = blockIdx.x, d = threadIdx.x;
    __shared__ float red[256], lgt[8], att[8], pl[256], ph[128], lo3[3];
    const float* tb = total + (size_t)b * 2048;

    for (int l = 0; l < 8; ++l) {
        red[d] = tb[l*256 + d] * poolw[d];
        __syncthreads();
        for (int s = 128; s > 0; s >>= 1) { if (d < s) red[d] += red[d + s]; __syncthreads(); }
        if (d == 0) lgt[l] = red[0] + poolb[0];
        __syncthreads();
    }
    if (d == 0) {
        float lg[8]; float m = -INFINITY;
        for (int l = 0; l < 8; ++l) {
            bool pad = false;
            if (l > 0)
                pad = (((jetRaw[(size_t)b*42 + (l-1)*6] - 63.964f) / 50.1006f) == 0.f);
            lg[l] = pad ? -INFINITY : lgt[l];
            m = fmaxf(m, lg[l]);
        }
        float sum = 0.f;
        for (int l = 0; l < 8; ++l) { att[l] = __expf(lg[l] - m); sum += att[l]; }
        for (int l = 0; l < 8; ++l) att[l] /= sum;
    }
    __syncthreads();

    float pooled = 0.f;
    for (int l = 0; l < 8; ++l) pooled += att[l] * tb[l*256 + d];
    pl[d] = pooled; __syncthreads();

    if (d < 128) {
        const float* w = ppw + (size_t)d * 256;
        float a = ppb[d];
        #pragma unroll 8
        for (int k = 0; k < 256; ++k) a += pl[k] * w[k];
        ph[d] = a / (1.f + __expf(-a));
    }
    __syncthreads();
    if (d < 3) {
        const float* w = ow + (size_t)d * 128;
        float a = ob[d];
        #pragma unroll 8
        for (int k = 0; k < 128; ++k) a += ph[k] * w[k];
        lo3[d] = a;
    }
    __syncthreads();
    if (d == 0) {
        const float m = fmaxf(lo3[0], fmaxf(lo3[1], lo3[2]));
        const float s = __expf(lo3[0]-m) + __expf(lo3[1]-m) + __expf(lo3[2]-m);
        const float lse = m + __logf(s);
        out[(size_t)b*3 + 0] = lo3[0] - lse;
        out[(size_t)b*3 + 1] = lo3[1] - lse;
        out[(size_t)b*3 + 2] = lo3[2] - lse;
    }
}

// ---------------------------------------------------------------------------
extern "C" void kernel_launch(void* const* d_in, const int* in_sizes, int n_in,
                              void* d_out, int out_size, void* d_ws, size_t ws_size,
                              hipStream_t stream)
{
    (void)in_sizes; (void)n_in; (void)out_size; (void)ws_size;
    const size_t B = BATCH;

    const float* mu     = (const float*)d_in[0];
    const float* nu     = (const float*)d_in[1];
    const float* jetR   = (const float*)d_in[2];
    const float* masses = (const float*)d_in[3];
    const float* mu1w = (const float*)d_in[4],  *mu1b = (const float*)d_in[5];
    const float* mu2w = (const float*)d_in[6],  *mu2b = (const float*)d_in[7];
    const float* nu1w = (const float*)d_in[8],  *nu1b = (const float*)d_in[9];
    const float* nu2w = (const float*)d_in[10], *nu2b = (const float*)d_in[11];
    const float* evw  = (const float*)d_in[12], *evb  = (const float*)d_in[13];
    const float* j1w  = (const float*)d_in[14], *j1b  = (const float*)d_in[15];
    const float* j2w  = (const float*)d_in[16], *j2b  = (const float*)d_in[17];
    const float* m1w  = (const float*)d_in[18], *m1b  = (const float*)d_in[19];
    const float* m2w  = (const float*)d_in[20], *m2b  = (const float*)d_in[21];
    const float* mng  = (const float*)d_in[22], *mnb  = (const float*)d_in[23];
    const float* ml1w = (const float*)d_in[24], *ml1b = (const float*)d_in[25];
    const float* ml2w = (const float*)d_in[26], *ml2b = (const float*)d_in[27];
    const float* jinw = (const float*)d_in[28], *jinb = (const float*)d_in[29];
    const float* joutw= (const float*)d_in[30], *joutb= (const float*)d_in[31];
    const float* jmlpw= (const float*)d_in[32], *jmlpb= (const float*)d_in[33];
    const float* ainw = (const float*)d_in[34], *ainb = (const float*)d_in[35];
    const float* aoutw= (const float*)d_in[36], *aoutb= (const float*)d_in[37];
    const float* amlpw= (const float*)d_in[38], *amlpb= (const float*)d_in[39];
    const float* prew = (const float*)d_in[40], *preb = (const float*)d_in[41];
    const float* poolw= (const float*)d_in[42], *poolb= (const float*)d_in[43];
    const float* ppw  = (const float*)d_in[44], *ppb  = (const float*)d_in[45];
    const float* ow   = (const float*)d_in[46], *ob   = (const float*)d_in[47];

    // ---- workspace carve-up (~418 MB) ----
    char* p = (char*)d_ws;
    auto alloc = [&](size_t bytes) -> void* {
        void* r = (void*)p;
        p += (bytes + 255) & ~(size_t)255;
        return r;
    };
    float*    evin    = (float*)   alloc(B * 256 * 4);
    float*    mass_h  = (float*)   alloc(B * 128 * 4);
    float*    jet_h   = (float*)   alloc(B * 7 * 128 * 4);
    float*    total   = (float*)   alloc(B * 8 * 256 * 4);
    float*    jetbuf  = (float*)   alloc(B * 7 * 256 * 4);
    float*    massbuf = (float*)   alloc(B * 256 * 4);
    float*    jm_raw  = (float*)   alloc(B * 28 * 4);
    float*    tm_raw  = (float*)   alloc(B * 36 * 4);
    _Float16* qkv     = (_Float16*)alloc((size_t)BCHUNK * 8 * 768 * 2);
    float*    ctx     = (float*)   alloc((size_t)BCHUNK * 8 * 256 * 4);
    float*    obuf    = (float*)   alloc((size_t)BCHUNK * 8 * 256 * 4);
    // f16 copies of GEMM weights (~1.7 MB)
    _Float16* evwh    = (_Float16*)alloc(65536 * 2);
    _Float16* j2wh    = (_Float16*)alloc(32768 * 2);
    _Float16* m2wh    = (_Float16*)alloc(32768 * 2);
    _Float16* jinwh   = (_Float16*)alloc(196608 * 2);
    _Float16* joutwh  = (_Float16*)alloc(65536 * 2);
    _Float16* jmlpwh  = (_Float16*)alloc(65536 * 2);
    _Float16* ainwh   = (_Float16*)alloc(196608 * 2);
    _Float16* aoutwh  = (_Float16*)alloc(65536 * 2);
    _Float16* amlpwh  = (_Float16*)alloc(65536 * 2);
    _Float16* prewh   = (_Float16*)alloc(65536 * 2);

    // 0) convert GEMM weights to f16 once
    auto cvt = [&](const float* s, _Float16* d, int n) {
        k_cvt<<<(n + 255) / 256, 256, 0, stream>>>(s, d, n);
    };
    cvt(evw,   evwh,   65536);
    cvt(j2w,   j2wh,   32768);
    cvt(m2w,   m2wh,   32768);
    cvt(jinw,  jinwh,  196608);
    cvt(joutw, joutwh, 65536);
    cvt(jmlpw, jmlpwh, 65536);
    cvt(ainw,  ainwh,  196608);
    cvt(aoutw, aoutwh, 65536);
    cvt(amlpw, amlpwh, 65536);
    cvt(prew,  prewh,  65536);

    // 1) normalization + tiny layers
    k_embed<<<(int)B, 128, 0, stream>>>(mu, nu, jetR, masses,
        mu1w, mu1b, mu2w, mu2b, nu1w, nu1b, nu2w, nu2b,
        m1w, m1b, j1w, j1b, evin, mass_h, jet_h);

    // 2) ev_mlp 256->256 (silu) -> total token 0
    k_gemm<256,1,false,1><<<(int)(B/128), 128, 0, stream>>>(
        evin, evwh, evb, 256, total, nullptr, 0, 0, 2048);
    // 3) jet layer2 128->256 (silu) -> jetbuf [B*7,256]
    k_gemm<128,1,false,1><<<(int)(B*7/128), 128, 0, stream>>>(
        jet_h, j2wh, j2b, 256, jetbuf, nullptr, 0, 0, 256);
    // 4) mass layer2 128->256 (silu) -> massbuf, then layernorm + linears
    k_gemm<128,1,false,1><<<(int)(B/128), 128, 0, stream>>>(
        mass_h, m2wh, m2b, 256, massbuf, nullptr, 0, 0, 256);
    k_lnorm<<<(int)B, 256, 0, stream>>>(massbuf, mng, mnb);
    k_masslin<<<(int)B, 64, 0, stream>>>(massbuf, ml1w, ml1b, ml2w, ml2b, jm_raw, tm_raw);

    // 5) jet attention block, chunked over batch
    for (int c = 0; c < NCHUNK; ++c) {
        const int c0 = c * BCHUNK;
        k_gemm<256,0,true,1><<<(BCHUNK*7)/128, 128, 0, stream>>>(
            jetbuf + (size_t)c0*7*256, jinwh, jinb, 768, nullptr, qkv, 0, 0, 768);
        k_attn<7><<<(BCHUNK*14 + 255)/256, 256, 0, stream>>>(
            qkv, jm_raw, jetR, ctx, c0, BCHUNK);
        k_gemm<256,0,false,1><<<(BCHUNK*7)/128, 128, 0, stream>>>(
            ctx, joutwh, joutb, 256, obuf, nullptr, 0, 0, 256);
        k_gemm<256,1,false,7><<<(BCHUNK*7)/128, 128, 0, stream>>>(
            obuf, jmlpwh, jmlpb, 256, total + (size_t)c0*2048, nullptr, 1, 256, 2048);
    }

    // 6) pre_mlp (silu + nan_to_num), in place on total [B*8,256]
    k_gemm<256,2,false,1><<<(int)(B*8/128), 128, 0, stream>>>(
        total, prewh, preb, 256, total, nullptr, 0, 0, 256);

    // 7) all-token attention block, chunked
    for (int c = 0; c < NCHUNK; ++c) {
        const int c0 = c * BCHUNK;
        k_gemm<256,0,true,1><<<(BCHUNK*8)/128, 128, 0, stream>>>(
            total + (size_t)c0*2048, ainwh, ainb, 768, nullptr, qkv, 0, 0, 768);
        k_attn<8><<<(BCHUNK*16 + 255)/256, 256, 0, stream>>>(
            qkv, tm_raw, jetR, ctx, c0, BCHUNK);
        k_gemm<256,0,false,1><<<(BCHUNK*8)/128, 128, 0, stream>>>(
            ctx, aoutwh, aoutb, 256, obuf, nullptr, 0, 0, 256);
        k_gemm<256,2,false,1><<<(BCHUNK*8)/128, 128, 0, stream>>>(
            obuf, amlpwh, amlpb, 256, total + (size_t)c0*2048, nullptr, 0, 0, 256);
    }

    // 8) pooled attention + head
    k_head<<<(int)B, 256, 0, stream>>>(total, jetR, poolw, poolb, ppw, ppb, ow, ob,
                                       (float*)d_out);
}
